// Encoder_16003048145604
// MI455X (gfx1250) — matrix operand order repaired and checked by reading, not compiled, not run on hardware
//
#include <hip/hip_runtime.h>
#include <math.h>

// ---- problem dims ----
#define Vv   1024
#define Hh   512
#define Bb   32
#define Tt   256
#define N3H  1536   // 3*H

typedef __attribute__((ext_vector_type(16))) __bf16 v16bf;
typedef __attribute__((ext_vector_type(8)))  __bf16 v8bf;
typedef __attribute__((ext_vector_type(8)))  float  v8f;

__device__ __forceinline__ v16bf combine16(v8bf lo, v8bf hi) {
    v16bf r;
#pragma unroll
    for (int e = 0; e < 8; ++e) { r[e] = lo[e]; r[e + 8] = hi[e]; }
    return r;
}

__device__ __forceinline__ v8f wmma_bf16(v16bf a, v16bf b, v8f c) {
    // D = A(16x32 bf16) * B(32x16 bf16) + C(16x16 f32)
    return __builtin_amdgcn_wmma_f32_16x16x32_bf16(
        /*neg_a=*/false, a, /*neg_b=*/false, b,
        /*c_mod=*/(short)0, c, /*reuse_a=*/false, /*reuse_b=*/false);
}

__device__ __forceinline__ float sigmoidf_(float x) {
    return 1.0f / (1.0f + __expf(-x));
}

// -----------------------------------------------------------------------------
// Pack a [512,1536] f32 matrix into bf16 K-tile column-major layout:
//   dst[(kt*1536 + n)*32 + kk] = src[(kt*32+kk)*1536 + n]
// so one B fragment = 2x contiguous 16B loads per lane.
// -----------------------------------------------------------------------------
__global__ void pack_w_kernel(const float* __restrict__ src, __bf16* __restrict__ dst) {
    int tid = blockIdx.x * blockDim.x + threadIdx.x;
    if (tid >= Hh * N3H) return;
    int kk = tid & 31;
    int n  = (tid >> 5) % N3H;
    int kt = tid / (32 * N3H);
    dst[tid] = (__bf16)src[(size_t)(kt * 32 + kk) * N3H + n];
}

// -----------------------------------------------------------------------------
// GRU recurrent scan. grid.x = direction (0 fwd, 1 bwd). 1024 threads = 32 waves
// (8 waves/SIMD: inter-wave interleave hides the L2 latency of the B loads —
// deliberately NO intra-wave multi-fragment buffering, it spills at this
// occupancy). Wave w owns hidden columns [w*16, w*16+16); computes z/r/g tiles
// for those columns (n, 512+n, 1024+n) for both M-tiles -> 6 WMMA accs.
// h state lives in LDS as bf16 [32][512].
// -----------------------------------------------------------------------------
__global__ __launch_bounds__(1024) void gru_scan_kernel(
    int layer,
    const int*   __restrict__ tokens,
    const float* __restrict__ W0f_, const float* __restrict__ W0b_,     // embedding (layer 0)
    const float* __restrict__ biasf_, const float* __restrict__ biasb_, // [2,1536]
    const __bf16* __restrict__ Upf_, const __bf16* __restrict__ Upb_,   // packed U
    const __bf16* __restrict__ xzf_, const __bf16* __restrict__ xzb_,   // layer-1 xz (bf16)
    __bf16* __restrict__ x0f_, __bf16* __restrict__ x0b_,               // layer-0 output (bf16)
    float* __restrict__ out,                                            // final output [B,T,2H]
    float* __restrict__ sdst)                                           // finals [B,2H]
{
    const int dir = blockIdx.x;
    const float*  W0   = dir ? W0b_  : W0f_;
    const float*  bias = dir ? biasb_ : biasf_;
    const __bf16* Up   = dir ? Upb_  : Upf_;
    const __bf16* xz   = dir ? xzb_  : xzf_;
    __bf16*       x0   = dir ? x0b_  : x0f_;

    __shared__ __bf16 hsh[Bb * Hh];   // 32 KB

    const int tid  = threadIdx.x;
    const int wave = tid >> 5;
    const int lane = tid & 31;
    const int half = lane >> 4;       // 0/1
    const int lm   = lane & 15;
    const int col  = wave * 16 + lm;  // hidden column this lane-slot owns

    // recurrent biases bh = bias[1, :] are step-invariant: hoist
    const float bh_g0 = bias[N3H + 0 * Hh + col];
    const float bh_g1 = bias[N3H + 1 * Hh + col];
    const float bh_g2 = bias[N3H + 2 * Hh + col];
    // layer-0 input biases b[0, :] (unused for layer 1)
    float bi_g[3] = {0.f, 0.f, 0.f};
    if (layer == 0) {
#pragma unroll
        for (int g = 0; g < 3; ++g) bi_g[g] = bias[g * Hh + col];
    }

    // per-gate B base pointers; kt advances by N3H*32 elems (uniform SGPR add)
    const __bf16* bbase[3];
#pragma unroll
    for (int g = 0; g < 3; ++g)
        bbase[g] = Up + ((size_t)(g * Hh + col)) * 32 + half * 8;

    // h0 = 0
    for (int i = tid; i < Bb * Hh; i += blockDim.x) hsh[i] = (__bf16)0.0f;
    __syncthreads();

    for (int t = 0; t < Tt; ++t) {
        v8f acc[2][3];
#pragma unroll
        for (int mt = 0; mt < 2; ++mt)
#pragma unroll
            for (int g = 0; g < 3; ++g)
#pragma unroll
                for (int i = 0; i < 8; ++i) acc[mt][g][i] = 0.0f;

        // inner = h @ U  (K = 512 in 16 k-tiles of 32)
        for (int kt = 0; kt < 16; ++kt) {
            const int kb = kt * 32 + half * 8;
            v16bf a0 = combine16(*(const v8bf*)&hsh[lm * Hh + kb],
                                 *(const v8bf*)&hsh[lm * Hh + kb + 16]);
            v16bf a1 = combine16(*(const v8bf*)&hsh[(16 + lm) * Hh + kb],
                                 *(const v8bf*)&hsh[(16 + lm) * Hh + kb + 16]);
            const size_t ko = (size_t)kt * (N3H * 32);
#pragma unroll
            for (int g = 0; g < 3; ++g) {
                const __bf16* bp = bbase[g] + ko;
                v16bf bfrag = combine16(*(const v8bf*)bp, *(const v8bf*)(bp + 16));
                acc[0][g] = wmma_bf16(a0, bfrag, acc[0][g]);
                acc[1][g] = wmma_bf16(a1, bfrag, acc[1][g]);
            }
        }
        __syncthreads();  // all h reads of step t complete

        // gates + state update; each lane owns (row, col) pairs exclusively
#pragma unroll
        for (int mt = 0; mt < 2; ++mt) {
#pragma unroll
            for (int i = 0; i < 8; ++i) {
                const int row = mt * 16 + half * 8 + i;  // batch index
                const float hold = (float)hsh[row * Hh + col];

                float xzv[3];
                if (layer == 0) {
                    const int tt_ = dir ? (Tt - 1 - t) : t;
                    const int tk = tokens[row * Tt + tt_];
#pragma unroll
                    for (int g = 0; g < 3; ++g)
                        xzv[g] = W0[(size_t)tk * N3H + g * Hh + col] + bi_g[g];
                    if (t + 1 < Tt) {   // prefetch next step's embedding rows
                        const int tt2 = dir ? (Tt - 2 - t) : (t + 1);
                        const int tk2 = tokens[row * Tt + tt2];
#pragma unroll
                        for (int g = 0; g < 3; ++g)
                            __builtin_prefetch(&W0[(size_t)tk2 * N3H + g * Hh + col], 0, 1);
                    }
                } else {
#pragma unroll
                    for (int g = 0; g < 3; ++g)
                        xzv[g] = (float)xz[((size_t)(row * Tt + t)) * N3H + g * Hh + col];
                    if (t + 1 < Tt) {   // prefetch next step's xz (streams from HBM)
#pragma unroll
                        for (int g = 0; g < 3; ++g)
                            __builtin_prefetch(&xz[((size_t)(row * Tt + t + 1)) * N3H + g * Hh + col], 0, 1);
                    }
                }

                const float z  = sigmoidf_(xzv[0] + acc[mt][0][i] + bh_g0);
                const float r  = sigmoidf_(xzv[1] + acc[mt][1][i] + bh_g1);
                const float hh = tanhf(xzv[2] + r * (acc[mt][2][i] + bh_g2));
                const float hn = z * hold + (1.0f - z) * hh;

                hsh[row * Hh + col] = (__bf16)hn;

                if (layer == 0) {
                    x0[((size_t)(row * Tt + t)) * Hh + col] = (__bf16)hn;
                } else {
                    const int tout = dir ? (Tt - 1 - t) : t;
                    out[((size_t)(row * Tt + tout)) * (2 * Hh) + dir * Hh + col] = hn;
                }
                if (t == Tt - 1)
                    sdst[row * (2 * Hh) + dir * Hh + col] = hn;
            }
        }
        __syncthreads();  // new h visible before next step's A loads
    }
}

// -----------------------------------------------------------------------------
// Layer-1 input GEMM: xz1[b*T+t][n] = x0[b*T+t][:] . W1[:][n] + b1[0][n]
// M=8192, N=1536, K=512. Block = 8 waves; wave -> 32x16 strip (2 M-tiles).
// A and B double-buffered across k (plenty of VGPRs at 8 waves/block).
// grid = (M/32, N/128, 2 dirs)
// -----------------------------------------------------------------------------
__global__ __launch_bounds__(256) void input_gemm_kernel(
    const __bf16* __restrict__ x0f_, const __bf16* __restrict__ x0b_,
    const __bf16* __restrict__ Wpf_, const __bf16* __restrict__ Wpb_,
    const float*  __restrict__ b1f_, const float*  __restrict__ b1b_,
    __bf16* __restrict__ xzf_, __bf16* __restrict__ xzb_)
{
    const int dir = blockIdx.z;
    const __bf16* A   = dir ? x0b_ : x0f_;
    const __bf16* Wp  = dir ? Wpb_ : Wpf_;
    const float*  bia = dir ? b1b_ : b1f_;
    __bf16*       dst = dir ? xzb_ : xzf_;

    const int tid  = threadIdx.x;
    const int wave = tid >> 5;
    const int lane = tid & 31;
    const int half = lane >> 4;
    const int lm   = lane & 15;
    const int row0 = blockIdx.x * 32;
    const int n    = blockIdx.y * 128 + wave * 16 + lm;

    v8f acc[2];
#pragma unroll
    for (int mt = 0; mt < 2; ++mt)
#pragma unroll
        for (int i = 0; i < 8; ++i) acc[mt][i] = 0.0f;

    auto loadAB = [&](int kt, v16bf* a, v16bf* b) {
        const int kb = kt * 32 + half * 8;
        a[0] = combine16(*(const v8bf*)&A[(size_t)(row0 + lm) * Hh + kb],
                         *(const v8bf*)&A[(size_t)(row0 + lm) * Hh + kb + 16]);
        a[1] = combine16(*(const v8bf*)&A[(size_t)(row0 + 16 + lm) * Hh + kb],
                         *(const v8bf*)&A[(size_t)(row0 + 16 + lm) * Hh + kb + 16]);
        const __bf16* bp = Wp + ((size_t)kt * N3H + n) * 32 + half * 8;
        b[0] = combine16(*(const v8bf*)bp, *(const v8bf*)(bp + 16));
    };

    v16bf aA[2], bA[1], aB[2], bB[1];
    loadAB(0, aA, bA);
#pragma unroll
    for (int kt = 0; kt < 16; kt += 2) {
        loadAB(kt + 1, aB, bB);
        acc[0] = wmma_bf16(aA[0], bA[0], acc[0]);
        acc[1] = wmma_bf16(aA[1], bA[0], acc[1]);
        if (kt + 2 < 16) loadAB(kt + 2, aA, bA);
        acc[0] = wmma_bf16(aB[0], bB[0], acc[0]);
        acc[1] = wmma_bf16(aB[1], bB[0], acc[1]);
    }

    const float bv = bia[n];  // b1[0][n]
#pragma unroll
    for (int mt = 0; mt < 2; ++mt)
#pragma unroll
        for (int i = 0; i < 8; ++i) {
            const int row = row0 + mt * 16 + half * 8 + i;
            dst[(size_t)row * N3H + n] = (__bf16)(acc[mt][i] + bv);
        }
}

// -----------------------------------------------------------------------------
extern "C" void kernel_launch(void* const* d_in, const int* in_sizes, int n_in,
                              void* d_out, int out_size, void* d_ws, size_t ws_size,
                              hipStream_t stream) {
    const int*   tokens = (const int*)  d_in[0];
    const float* W0f = (const float*)d_in[1];
    const float* U0f = (const float*)d_in[2];
    const float* b0f = (const float*)d_in[3];
    const float* W1f = (const float*)d_in[4];
    const float* U1f = (const float*)d_in[5];
    const float* b1f = (const float*)d_in[6];
    const float* W0b = (const float*)d_in[7];
    const float* U0b = (const float*)d_in[8];
    const float* b0b = (const float*)d_in[9];
    const float* W1b = (const float*)d_in[10];
    const float* U1b = (const float*)d_in[11];
    const float* b1b = (const float*)d_in[12];

    float* out = (float*)d_out;                         // [B,T,2H]
    float* s0  = out + (size_t)Bb * Tt * 2 * Hh;        // [B,2H]
    float* s1  = s0 + (size_t)Bb * 2 * Hh;              // [B,2H]

    // workspace layout (bf16 elements)
    const size_t PACK = (size_t)Hh * N3H;               // 786432
    __bf16* p = (__bf16*)d_ws;
    __bf16* U0fp = p; p += PACK;
    __bf16* U1fp = p; p += PACK;
    __bf16* W1fp = p; p += PACK;
    __bf16* U0bp = p; p += PACK;
    __bf16* U1bp = p; p += PACK;
    __bf16* W1bp = p; p += PACK;
    __bf16* x0fp = p; p += (size_t)Bb * Tt * Hh;
    __bf16* x0bp = p; p += (size_t)Bb * Tt * Hh;
    __bf16* xz1f = p; p += (size_t)Bb * Tt * N3H;
    __bf16* xz1b = p; p += (size_t)Bb * Tt * N3H;

    // 1) pack weights to bf16 K-tile layout
    const int pthreads = 256;
    const int pblocks  = (int)((PACK + pthreads - 1) / pthreads);
    pack_w_kernel<<<pblocks, pthreads, 0, stream>>>(U0f, U0fp);
    pack_w_kernel<<<pblocks, pthreads, 0, stream>>>(U1f, U1fp);
    pack_w_kernel<<<pblocks, pthreads, 0, stream>>>(W1f, W1fp);
    pack_w_kernel<<<pblocks, pthreads, 0, stream>>>(U0b, U0bp);
    pack_w_kernel<<<pblocks, pthreads, 0, stream>>>(U1b, U1bp);
    pack_w_kernel<<<pblocks, pthreads, 0, stream>>>(W1b, W1bp);

    // 2) layer-0 scans, both directions concurrently (1 WGP each)
    gru_scan_kernel<<<2, 1024, 0, stream>>>(
        0, tokens, W0f, W0b, b0f, b0b, U0fp, U0bp,
        (const __bf16*)nullptr, (const __bf16*)nullptr,
        x0fp, x0bp, out, s0);

    // 3) layer-1 input GEMM (fully parallel WMMA GEMM)
    input_gemm_kernel<<<dim3(Bb * Tt / 32, N3H / 128, 2), 256, 0, stream>>>(
        x0fp, x0bp, W1fp, W1bp, b1f, b1b, xz1f, xz1b);

    // 4) layer-1 scans -> final output + s1
    gru_scan_kernel<<<2, 1024, 0, stream>>>(
        1, tokens, W0f, W0b, b1f, b1b, U1fp, U1bp,
        xz1f, xz1b, x0fp, x0bp, out, s1);
}
